// MoEMLP_72258529788599
// MI455X (gfx1250) — matrix operand order repaired and checked by reading, compiled, not run on hardware
//
#include <hip/hip_runtime.h>
#include <stdint.h>

#define DIMD 2048
#define HID  8192
#define NTOK 4096

#define TM 128
#define TN 128
#define TK 32
#define SA 40   // LDS row stride in 16-bit units (32 + 8 pad, keeps 16B alignment)
#define SB 40

typedef __attribute__((ext_vector_type(16))) __bf16 v16bf;
typedef __attribute__((ext_vector_type(8)))  float  v8f;

union FragBF { uint4 q[2]; v16bf b; };

__device__ __forceinline__ unsigned short f2bf(float f) {
  union { float f; unsigned int u; } v; v.f = f;
  unsigned int r = v.u + 0x7FFFu + ((v.u >> 16) & 1u);  // round-to-nearest-even
  return (unsigned short)(r >> 16);
}

// gfx1250 async global->LDS copy (ASYNCcnt-tracked), per-lane 16B.
__device__ __forceinline__ void async_b128(unsigned int lds_off, const void* g) {
  asm volatile("global_load_async_to_lds_b128 %0, %1, off"
               :: "v"(lds_off), "v"(g) : "memory");
}
__device__ __forceinline__ void wait_async0() {
  asm volatile("s_wait_asynccnt 0x0" ::: "memory");
}
// Flat LDS-aperture addresses carry the LDS byte offset in the low 32 bits.
__device__ __forceinline__ unsigned int lds_off_u32(const void* p) {
  return (unsigned int)(uint64_t)p;
}

// ---------------- fp32 -> bf16 bulk convert ----------------
__global__ void cvt_bf16_kernel(const float* __restrict__ src,
                                unsigned short* __restrict__ dst, int n4) {
  int i = blockIdx.x * blockDim.x + threadIdx.x;
  if (i >= n4) return;
  float4 f = ((const float4*)src)[i];
  uint2 o;
  o.x = (unsigned)f2bf(f.x) | ((unsigned)f2bf(f.y) << 16);
  o.y = (unsigned)f2bf(f.z) | ((unsigned)f2bf(f.w) << 16);
  ((uint2*)dst)[i] = o;
}

// ---------------- router: argmax expert + softmax gate, compact slots ----------------
__global__ void router_kernel(const float* __restrict__ x,
                              const float* __restrict__ wr,
                              int* __restrict__ idx_list,
                              float* __restrict__ tw_slot,
                              int* __restrict__ cnt) {
  const int lane = threadIdx.x & 31;
  const int wave = threadIdx.x >> 5;
  const int t = blockIdx.x * 8 + wave;
  const float* xr = x + (size_t)t * DIMD;
  float s0 = 0.f, s1 = 0.f;
  for (int i = lane; i < DIMD; i += 32) {
    float xv = xr[i];
    s0 += xv * wr[i];
    s1 += xv * wr[DIMD + i];
  }
#pragma unroll
  for (int m = 16; m >= 1; m >>= 1) {
    s0 += __shfl_xor(s0, m, 32);
    s1 += __shfl_xor(s1, m, 32);
  }
  if (lane == 0) {
    int e = (s1 > s0) ? 1 : 0;                         // argmax, first-index tie-break
    float w = 1.f / (1.f + __expf(-fabsf(s1 - s0)));   // softmax prob of the winner
    int slot = (e == 0) ? atomicAdd(&cnt[0], 1)
                        : (NTOK - 1 - atomicAdd(&cnt[1], 1));
    idx_list[slot] = t;
    tw_slot[slot] = w;
  }
}

// ---- shared tile helpers -------------------------------------------------

// Fragment loads (contiguous k = 16*half + s for A and B) + 8 WMMAs for one 128x128 block-tile stage.
__device__ __forceinline__ void tile_mma(const unsigned short* __restrict__ la,
                                         const unsigned short* __restrict__ lb,
                                         int wm, int wn, int half, int l16,
                                         v8f (&acc)[2][4]) {
  FragBF af[2], bf[4];
#pragma unroll
  for (int mi = 0; mi < 2; mi++) {
    const unsigned short* base = &la[(wm * 32 + mi * 16 + l16) * SA + half * 16];
    af[mi].q[0] = *(const uint4*)base;          // K = half*16 + {0..7}
    af[mi].q[1] = *(const uint4*)(base + 8);    // K = half*16 + {8..15}
  }
#pragma unroll
  for (int ni = 0; ni < 4; ni++) {
    const unsigned short* base = &lb[(wn * 64 + ni * 16 + l16) * SB + half * 16];
    bf[ni].q[0] = *(const uint4*)base;          // K = half*16 + {0..7}
    bf[ni].q[1] = *(const uint4*)(base + 8);    // K = half*16 + {8..15}
  }
#pragma unroll
  for (int mi = 0; mi < 2; mi++)
#pragma unroll
    for (int ni = 0; ni < 4; ni++)
      acc[mi][ni] = __builtin_amdgcn_wmma_f32_16x16x32_bf16(
          false, af[mi].b, false, bf[ni].b, (short)0, acc[mi][ni], false, false);
}

// Async-stream one 128x32 bf16 tile (row stride = elems, 64B/row) into LDS.
__device__ __forceinline__ void async_tile(const unsigned short* __restrict__ gbase,
                                           size_t row_stride, int k0,
                                           unsigned short* __restrict__ lbuf, int tid) {
#pragma unroll
  for (int j = 0; j < 2; j++) {
    int c = tid * 2 + j;
    int row = c >> 2;
    int part = c & 3;
    async_b128(lds_off_u32(&lbuf[row * SB + part * 8]),
               gbase + (size_t)row * row_stride + k0 + part * 8);
  }
}

// Gathered fp32 x row segment -> cvt -> packed bf16 LDS store (A tile for GEMM1).
__device__ __forceinline__ void cvt_tile_a(const float* __restrict__ aptr, int k0,
                                           unsigned short* __restrict__ adst) {
  const float4* p4 = (const float4*)(aptr + k0);
  float fv[16];
#pragma unroll
  for (int q = 0; q < 4; q++) {
    float4 f = p4[q];
    fv[q * 4 + 0] = f.x; fv[q * 4 + 1] = f.y;
    fv[q * 4 + 2] = f.z; fv[q * 4 + 3] = f.w;
  }
  unsigned int u[8];
#pragma unroll
  for (int q = 0; q < 8; q++)
    u[q] = (unsigned)f2bf(fv[2 * q]) | ((unsigned)f2bf(fv[2 * q + 1]) << 16);
  uint4* d4 = (uint4*)adst;
  d4[0] = make_uint4(u[0], u[1], u[2], u[3]);
  d4[1] = make_uint4(u[4], u[5], u[6], u[7]);
}

// ---------------- GEMM1: a = sq(leaky_relu(x @ w_fc[e]^T)), gathered tokens ----------------
__global__ __launch_bounds__(256)
void moe_fc_kernel(const float* __restrict__ x,
                   const unsigned short* __restrict__ wfc,   // bf16 [E][H][D]
                   const int* __restrict__ idx_list,
                   const int* __restrict__ cnt,
                   unsigned short* __restrict__ a_out) {     // bf16 [N][H] (slot-major)
  __shared__ __align__(16) unsigned short lds_a[2][TM * SA];
  __shared__ __align__(16) unsigned short lds_b[2][TN * SB];

  const int e  = blockIdx.z;
  const int c0 = cnt[0];
  const int lo = (e == 0) ? 0 : c0;
  const int hi = (e == 0) ? c0 : NTOK;
  const int s0 = blockIdx.y * TM;
  if (s0 >= hi || s0 + TM <= lo) return;           // tile fully outside this expert
  const int hbase = blockIdx.x * TN;

  const int tid  = threadIdx.x;
  const int lane = tid & 31;
  const int wave = tid >> 5;
  const int wm   = wave & 3;       // 4 wave rows x 32
  const int wn   = wave >> 2;      // 2 wave cols x 64
  const int half = lane >> 4;
  const int l16  = lane & 15;

  // A loader: each thread converts 16 fp32 of one gathered x row per K-step
  const int arow = tid >> 1;
  const int aseg = tid & 1;
  const int atok = idx_list[s0 + arow];
  const float* aptr = x + (size_t)atok * DIMD + aseg * 16;
  const int aoff = arow * SA + aseg * 16;

  const unsigned short* wbase = wfc + ((size_t)e * HID + hbase) * DIMD;

  v8f acc[2][4] = {};

  // --- software pipeline: prefetch stage 0, then overlap stage k+1 with MMA k ---
  async_tile(wbase, DIMD, 0, lds_b[0], tid);
  cvt_tile_a(aptr, 0, &lds_a[0][aoff]);
  wait_async0();
  __syncthreads();

  const int NK = DIMD / TK;
  for (int kt = 0; kt < NK; ++kt) {
    const int cur = kt & 1;
    if (kt + 1 < NK) {
      async_tile(wbase, DIMD, (kt + 1) * TK, lds_b[cur ^ 1], tid);
      cvt_tile_a(aptr, (kt + 1) * TK, &lds_a[cur ^ 1][aoff]);
    }
    tile_mma(lds_a[cur], lds_b[cur], wm, wn, half, l16, acc);
    wait_async0();        // next-stage async tiles landed
    __syncthreads();      // + ds stores visible, frag reads of 'cur' finished
  }

  // fused activation: sq(leaky_relu(h, 0.5)) -> bf16 intermediate
#pragma unroll
  for (int mi = 0; mi < 2; mi++) {
#pragma unroll
    for (int r = 0; r < 8; r++) {
      int slot = s0 + wm * 32 + mi * 16 + half * 8 + r;
      if (slot < lo || slot >= hi) continue;      // row belongs to the other expert
#pragma unroll
      for (int ni = 0; ni < 4; ni++) {
        int hcol = hbase + wn * 64 + ni * 16 + l16;
        float hv = acc[mi][ni][r];
        float vv = hv >= 0.f ? hv : 0.5f * hv;
        a_out[(size_t)slot * HID + hcol] = f2bf(vv * vv);
      }
    }
  }
}

// ---------------- GEMM2: out = gate * (a @ w_proj[e]^T), scatter to token rows ----------------
__global__ __launch_bounds__(256)
void moe_proj_kernel(const unsigned short* __restrict__ a_in,   // bf16 [N][H]
                     const unsigned short* __restrict__ wproj,  // bf16 [E][D][H]
                     const int* __restrict__ idx_list,
                     const float* __restrict__ tw_slot,
                     const int* __restrict__ cnt,
                     float* __restrict__ out) {
  __shared__ __align__(16) unsigned short lds_a[2][TM * SA];
  __shared__ __align__(16) unsigned short lds_b[2][TN * SB];

  const int e  = blockIdx.z;
  const int c0 = cnt[0];
  const int lo = (e == 0) ? 0 : c0;
  const int hi = (e == 0) ? c0 : NTOK;
  const int s0 = blockIdx.y * TM;
  if (s0 >= hi || s0 + TM <= lo) return;
  const int dbase = blockIdx.x * TN;

  const int tid  = threadIdx.x;
  const int lane = tid & 31;
  const int wave = tid >> 5;
  const int wm   = wave & 3;
  const int wn   = wave >> 2;
  const int half = lane >> 4;
  const int l16  = lane & 15;

  const unsigned short* abase = a_in + (size_t)s0 * HID;
  const unsigned short* wbase = wproj + ((size_t)e * DIMD + dbase) * HID;

  v8f acc[2][4] = {};

  // both operands already bf16 in global: pure async double-stream, double-buffered
  async_tile(abase, HID, 0, lds_a[0], tid);
  async_tile(wbase, HID, 0, lds_b[0], tid);
  wait_async0();
  __syncthreads();

  const int NK = HID / TK;
  for (int kt = 0; kt < NK; ++kt) {
    const int cur = kt & 1;
    if (kt + 1 < NK) {
      async_tile(abase, HID, (kt + 1) * TK, lds_a[cur ^ 1], tid);
      async_tile(wbase, HID, (kt + 1) * TK, lds_b[cur ^ 1], tid);
    }
    tile_mma(lds_a[cur], lds_b[cur], wm, wn, half, l16, acc);
    wait_async0();
    __syncthreads();
  }

#pragma unroll
  for (int mi = 0; mi < 2; mi++) {
#pragma unroll
    for (int r = 0; r < 8; r++) {
      int slot = s0 + wm * 32 + mi * 16 + half * 8 + r;
      if (slot < lo || slot >= hi) continue;
      int tok  = idx_list[slot];
      float gw = tw_slot[slot];
#pragma unroll
      for (int ni = 0; ni < 4; ni++) {
        int dcol = dbase + wn * 64 + ni * 16 + l16;
        out[(size_t)tok * DIMD + dcol] = acc[mi][ni][r] * gw;
      }
    }
  }
}

extern "C" void kernel_launch(void* const* d_in, const int* in_sizes, int n_in,
                              void* d_out, int out_size, void* d_ws, size_t ws_size,
                              hipStream_t stream) {
  const float* x        = (const float*)d_in[0];   // [4,1024,2048]
  const float* w_router = (const float*)d_in[1];   // [2,2048]
  const float* w_fc     = (const float*)d_in[2];   // [2,8192,2048]
  const float* w_proj   = (const float*)d_in[3];   // [2,2048,8192]
  float* out = (float*)d_out;

  // workspace layout
  char* ws = (char*)d_ws;
  const size_t SZ_WFC   = (size_t)2 * HID * DIMD * 2;  // 64 MiB bf16 w_fc
  const size_t SZ_WPROJ = (size_t)2 * DIMD * HID * 2;  // 64 MiB bf16 w_proj
  const size_t SZ_A     = (size_t)NTOK * HID * 2;      // 64 MiB bf16 activations
  unsigned short* wfc_b   = (unsigned short*)(ws);
  unsigned short* wproj_b = (unsigned short*)(ws + SZ_WFC);
  unsigned short* a_buf   = (unsigned short*)(ws + SZ_WFC + SZ_WPROJ);
  int*   idx_list = (int*)  (ws + SZ_WFC + SZ_WPROJ + SZ_A);
  float* tw_slot  = (float*)(ws + SZ_WFC + SZ_WPROJ + SZ_A + NTOK * 4);
  int*   cnt      = (int*)  (ws + SZ_WFC + SZ_WPROJ + SZ_A + NTOK * 8);

  hipMemsetAsync(cnt, 0, 2 * sizeof(int), stream);

  const int n4 = 2 * HID * DIMD / 4;  // same element count for both weights
  cvt_bf16_kernel<<<n4 / 256, 256, 0, stream>>>(w_fc, wfc_b, n4);
  cvt_bf16_kernel<<<n4 / 256, 256, 0, stream>>>(w_proj, wproj_b, n4);

  router_kernel<<<NTOK / 8, 256, 0, stream>>>(x, w_router, idx_list, tw_slot, cnt);

  moe_fc_kernel<<<dim3(HID / TN, NTOK / TM, 2), 256, 0, stream>>>(
      x, wfc_b, idx_list, cnt, a_buf);

  moe_proj_kernel<<<dim3(DIMD / TN, NTOK / TM, 2), 256, 0, stream>>>(
      a_buf, wproj_b, idx_list, tw_slot, cnt, out);

  (void)in_sizes; (void)n_in; (void)out_size; (void)ws_size;
}
